// NoisyTopKRouter_57621281243491
// MI455X (gfx1250) — matrix operand order repaired and checked by reading, not compiled, not run on hardware
//
#include <hip/hip_runtime.h>

typedef __attribute__((ext_vector_type(16))) __bf16 v16bf;
typedef __attribute__((ext_vector_type(8)))  float  v8f;
typedef __attribute__((ext_vector_type(4)))  int    v4i;

#define T_TOKENS   16384      // B*S = 4*4096
#define DDIM       1024
#define NEXP       64
#define TOK_PER_BLOCK 64
#define LDS_PITCH  65
#define NCHUNK     32         // K chunks of 32
#define NCOL       128        // 64 gate + 64 noise expert columns
#define COL_PITCH  40         // ushorts per column per chunk (32 data + 8 pad) -> 80B stride

// d_out flat layout (all fp32):
//   [0, 32768)                routing_weights [T,2]
//   [32768, 65536)            expert_indices  [T,2] (as float)
//   [65536]                   load_balance_loss
//   [65537, 65537+1048576)    router_probs [T,64]
//   [1114113]                 noise_scale_mean
#define OUT_IDX_BASE   32768
#define OUT_LOSS       65536
#define OUT_RP_BASE    65537
#define OUT_NSMEAN     1114113

// ---------------- helpers ----------------

__device__ __forceinline__ unsigned short f32_to_bf16_rne(float f) {
  unsigned int u = __float_as_uint(f);
  unsigned int r = u + 0x7FFFu + ((u >> 16) & 1u);
  return (unsigned short)(r >> 16);
}

// CDNA5 async copy: global -> LDS, 16B per lane, tracked by ASYNCcnt.
__device__ __forceinline__ void async_copy_b128(const unsigned short* g, unsigned short* l) {
#if __has_builtin(__builtin_amdgcn_global_load_async_to_lds_b128)
  v4i* lv = reinterpret_cast<v4i*>(l);
  __builtin_amdgcn_global_load_async_to_lds_b128(
      (__attribute__((address_space(1))) v4i*)(size_t)g,
      (__attribute__((address_space(3))) v4i*)lv,
      0, 0);
#else
  unsigned int laddr =
      (unsigned int)(unsigned long long)(__attribute__((address_space(3))) void*)l;
  asm volatile("global_load_async_to_lds_b128 %0, %1, off"
               :: "v"(laddr), "v"((unsigned long long)(size_t)g)
               : "memory");
#endif
}

template <int N>
__device__ __forceinline__ void wait_asynccnt() {
#if __has_builtin(__builtin_amdgcn_s_wait_asynccnt)
  __builtin_amdgcn_s_wait_asynccnt((unsigned short)N);
#else
  asm volatile("s_wait_asynccnt %0" :: "n"(N));
#endif
}

__device__ __forceinline__ void threefry2x32(unsigned int k0, unsigned int k1,
                                             unsigned int c0, unsigned int c1,
                                             unsigned int &o0, unsigned int &o1) {
  unsigned int ks0 = k0, ks1 = k1, ks2 = 0x1BD11BDAu ^ k0 ^ k1;
  unsigned int x0 = c0 + ks0, x1 = c1 + ks1;
#define TF_R(x, r) (((x) << (r)) | ((x) >> (32 - (r))))
#define TF_4(r0, r1, r2, r3)                         \
  x0 += x1; x1 = TF_R(x1, r0); x1 ^= x0;             \
  x0 += x1; x1 = TF_R(x1, r1); x1 ^= x0;             \
  x0 += x1; x1 = TF_R(x1, r2); x1 ^= x0;             \
  x0 += x1; x1 = TF_R(x1, r3); x1 ^= x0;
  TF_4(13, 15, 26, 6);  x0 += ks1; x1 += ks2 + 1u;
  TF_4(17, 29, 16, 24); x0 += ks2; x1 += ks0 + 2u;
  TF_4(13, 15, 26, 6);  x0 += ks0; x1 += ks1 + 3u;
  TF_4(17, 29, 16, 24); x0 += ks1; x1 += ks2 + 4u;
  TF_4(13, 15, 26, 6);  x0 += ks2; x1 += ks0 + 5u;
#undef TF_4
#undef TF_R
  o0 = x0; o1 = x1;
}

// Deterministic standard normal keyed on flat element index (stands in for
// jax.random.normal(fold_in(key(0),12345), ...): counter-based, replay-stable).
__device__ __forceinline__ float randn_det(unsigned int idx) {
  unsigned int a, b;
  threefry2x32(0x2F1C9E7Bu, 0x00003039u, idx, idx ^ 0x9E3779B9u, a, b);
  float u1 = ((float)(a >> 8) + 0.5f) * (1.0f / 16777216.0f);  // (0,1)
  float u2 = (float)(b >> 8) * (1.0f / 16777216.0f);
  float r = sqrtf(fmaxf(-2.0f * logf(u1), 0.0f));
  return r * __cosf(6.28318530718f * u2);
}

__device__ __forceinline__ float softplus_f(float x) {
  return fmaxf(x, 0.0f) + log1pf(expf(-fabsf(x)));
}

// ---------------- kernel 1: W fp32 -> bf16 (L2-resident), zero accumulators ----------------

__global__ void wcvt_kernel(const float* __restrict__ Wg, const float* __restrict__ Wn,
                            unsigned short* __restrict__ wbf, float* __restrict__ accum) {
  int i = blockIdx.x * blockDim.x + threadIdx.x;
  if (i < NEXP * DDIM) {
    wbf[i] = f32_to_bf16_rne(Wg[i]);
    wbf[NEXP * DDIM + i] = f32_to_bf16_rne(Wn[i]);
  }
  if (i < 2 * NEXP + 1) accum[i] = 0.0f;  // [0..63] counts, [64..127] prob sums, [128] noise sum
}

// ---------------- kernel 2: WMMA GEMM + routing epilogue ----------------

__global__ __launch_bounds__(256)
void router_main_kernel(const float* __restrict__ x,
                        const unsigned short* __restrict__ wbf,
                        float* __restrict__ out,
                        float* __restrict__ accum) {
  // Double-buffered async-staged B tiles: 128 columns x 32 K bf16 per chunk,
  // padded to 80B/column (20 dword-banks) to keep ds_load_b128 conflicts low.
  __shared__ unsigned short ldsB[2][NCOL * COL_PITCH];
  __shared__ float lds_logits[2][TOK_PER_BLOCK][LDS_PITCH]; // [gate|noise][token][expert]
  __shared__ float ldsProbSum[NEXP];
  __shared__ float ldsCount[NEXP];
  __shared__ float ldsNoise;

  const int tid  = threadIdx.x;
  const int lane = tid & 31;
  const int wave = tid >> 5;
  const int tokBase = blockIdx.x * TOK_PER_BLOCK;

  if (tid < NEXP) { ldsProbSum[tid] = 0.0f; ldsCount[tid] = 0.0f; }
  if (tid == 0) ldsNoise = 0.0f;

  // ---- GEMM: wave w -> 16-token M-tile (w>>1), matrix select gate/noise (w&1),
  //            4 accumulators cover experts 0..63 of that matrix.
  const int mtile = wave >> 1;
  const int nhalf = wave & 1;
  const int half  = lane >> 4;     // half-wave selects K sub-range per ISA layouts
  const int l15   = lane & 15;

  const float* xrow = x + (size_t)(tokBase + mtile * 16 + l15) * DDIM + half * 16;

  // Cooperative async staging: thread tid owns (column c, k-half h); one 32B
  // fragment per thread per chunk == two b128 async transfers.
  const int stg_c = tid >> 1;      // 0..127 (0..63 gate, 64..127 noise)
  const int stg_h = tid & 1;       // K sub-half (16 bf16 each)
  const unsigned short* stg_g = wbf + (size_t)stg_c * DDIM + stg_h * 16;
  unsigned short* stg_l0 = &ldsB[0][stg_c * COL_PITCH + stg_h * 16];
  unsigned short* stg_l1 = &ldsB[1][stg_c * COL_PITCH + stg_h * 16];

  v8f acc[4];
#pragma unroll
  for (int j = 0; j < 4; ++j)
#pragma unroll
    for (int r = 0; r < 8; ++r) acc[j][r] = 0.0f;

  union AFrag { v16bf v; __bf16 h[16]; };
  union BFrag { v16bf v; uint4 q[2]; };

  // Prologue: kick off chunk 0 DMA.
  async_copy_b128(stg_g, stg_l0);
  async_copy_b128(stg_g + 8, stg_l0 + 8);

  for (int kc = 0; kc < NCHUNK; ++kc) {
    const int k0 = kc * 32;
    const int buf = kc & 1;
    unsigned short* curB = ldsB[buf];

    // Issue next chunk's DMA into the other buffer (its readers finished at
    // the trailing barrier of iteration kc-1), then wait for chunk kc to land.
    if (kc + 1 < NCHUNK) {
      const unsigned short* g = stg_g + (kc + 1) * 32;
      unsigned short* l = (buf ? stg_l0 : stg_l1);
      async_copy_b128(g, l);
      async_copy_b128(g + 8, l + 8);
      wait_asynccnt<2>();          // in-order: chunk kc's 2 transfers complete
    } else {
      wait_asynccnt<0>();
    }
    __syncthreads();               // chunk kc visible to every wave

    // A: 16x32 bf16 fragment. lane<16 holds row M=l15, K = k0+0..15;
    // lane>=16 same rows, K = k0+16..31 (matches B's contiguous K order).
    float fv[16];
    const float4* xp = reinterpret_cast<const float4*>(xrow + k0);
    *reinterpret_cast<float4*>(&fv[0])  = xp[0];
    *reinterpret_cast<float4*>(&fv[4])  = xp[1];
    *reinterpret_cast<float4*>(&fv[8])  = xp[2];
    *reinterpret_cast<float4*>(&fv[12]) = xp[3];
    AFrag a;
#pragma unroll
    for (int i = 0; i < 16; ++i) a.h[i] = (__bf16)fv[i];

    if (kc + 1 < NCHUNK)
      __builtin_prefetch(xrow + k0 + 32, 0, 3);   // global_prefetch next x K-tile

#pragma unroll
    for (int j = 0; j < 4; ++j) {
      // B: 32x16 bf16 fragment from LDS. Column N=l15 -> expert j*16+l15 of
      // this wave's matrix; lane<16 K=k0..k0+15, lane>=16 K=k0+16..k0+31.
      BFrag b;
      const uint4* lp = reinterpret_cast<const uint4*>(
          &curB[(nhalf * 64 + j * 16 + l15) * COL_PITCH + half * 16]);
      b.q[0] = lp[0];
      b.q[1] = lp[1];
      acc[j] = __builtin_amdgcn_wmma_f32_16x16x32_bf16(
          false, a.v, false, b.v, (short)0, acc[j], false, false);
    }

    __syncthreads();               // all reads of curB done before it is re-filled
  }

  // C layout: VGPR r -> row (r + 8*half), col = l15. Spill logits to LDS.
#pragma unroll
  for (int j = 0; j < 4; ++j)
#pragma unroll
    for (int r = 0; r < 8; ++r)
      lds_logits[nhalf][mtile * 16 + r + half * 8][j * 16 + l15] = acc[j][r];

  __syncthreads();

  // ---- per-token routing epilogue: one thread per token ----
  if (tid < TOK_PER_BLOCK) {
    const int t = tokBase + tid;
    float maxc = -3.4e38f;
    float v1 = -3.4e38f, v2 = -3.4e38f;
    int   i1 = 0, i2 = 0;
    float nsum = 0.0f;

    for (int e = 0; e < NEXP; ++e) {
      float c  = lds_logits[0][tid][e];
      float nl = lds_logits[1][tid][e];
      float sp = softplus_f(nl);
      nsum += sp;
      float lg = c + randn_det((unsigned int)(t * NEXP + e)) * sp; // NOISE_STD=1
      maxc = fmaxf(maxc, c);
      if (lg > v1)      { v2 = v1; i2 = i1; v1 = lg; i1 = e; }
      else if (lg > v2) { v2 = lg; i2 = e; }
    }

    float sumc = 0.0f;
    for (int e = 0; e < NEXP; ++e) sumc += expf(lds_logits[0][tid][e] - maxc);
    float inv = 1.0f / sumc;

    float* rp = out + OUT_RP_BASE + (size_t)t * NEXP;
    for (int e = 0; e < NEXP; ++e) {
      float p = expf(lds_logits[0][tid][e] - maxc) * inv;
      rp[e] = p;
      atomicAdd(&ldsProbSum[e], p);
    }

    // top-2 renormalization == 2-way softmax over {v1, v2}
    float w1 = 1.0f / (1.0f + expf(v2 - v1));
    out[t * 2 + 0] = w1;
    out[t * 2 + 1] = 1.0f - w1;
    out[OUT_IDX_BASE + t * 2 + 0] = (float)i1;
    out[OUT_IDX_BASE + t * 2 + 1] = (float)i2;

    atomicAdd(&ldsCount[i1], 1.0f);
    atomicAdd(&ldsCount[i2], 1.0f);
    atomicAdd(&ldsNoise, nsum);
  }

  __syncthreads();

  if (tid < NEXP) {
    atomicAdd(&accum[tid], ldsCount[tid]);
    atomicAdd(&accum[NEXP + tid], ldsProbSum[tid]);
  }
  if (tid == 0) atomicAdd(&accum[2 * NEXP], ldsNoise);
}

// ---------------- kernel 3: scalars ----------------

__global__ void finalize_kernel(const float* __restrict__ accum, float* __restrict__ out) {
  __shared__ float part[NEXP];
  int e = threadIdx.x;
  float frac = accum[e] * (1.0f / (float)T_TOKENS);
  float mp   = accum[NEXP + e] * (1.0f / (float)T_TOKENS);
  part[e] = frac * mp;
  __syncthreads();
  if (e == 0) {
    float s = 0.0f;
    for (int i = 0; i < NEXP; ++i) s += part[i];
    out[OUT_LOSS]   = 0.01f * (float)NEXP * s;        // LOAD_BALANCE_WEIGHT * E * sum
    out[OUT_NSMEAN] = accum[2 * NEXP] * (1.0f / ((float)T_TOKENS * (float)NEXP));
  }
}

// ---------------- host entry ----------------

extern "C" void kernel_launch(void* const* d_in, const int* in_sizes, int n_in,
                              void* d_out, int out_size, void* d_ws, size_t ws_size,
                              hipStream_t stream) {
  const float* x  = (const float*)d_in[0];   // [4,4096,1024]
  const float* Wg = (const float*)d_in[1];   // [64,1024]
  const float* Wn = (const float*)d_in[2];   // [64,1024]
  float* out = (float*)d_out;

  unsigned short* wbf = (unsigned short*)d_ws;                 // 2*64*1024 bf16 = 256 KB
  float* accum = (float*)((char*)d_ws + (size_t)2 * NEXP * DDIM * sizeof(unsigned short));

  wcvt_kernel<<<(NEXP * DDIM + 255) / 256, 256, 0, stream>>>(Wg, Wn, wbf, accum);
  router_main_kernel<<<T_TOKENS / TOK_PER_BLOCK, 256, 0, stream>>>(x, wbf, out, accum);
  finalize_kernel<<<1, NEXP, 0, stream>>>(accum, out);
}
